// SAModule_30631706755377
// MI455X (gfx1250) — matrix operand order repaired and checked by reading, not compiled, not run on hardware
//
#include <hip/hip_runtime.h>
#include <hip/hip_bf16.h>

typedef __attribute__((ext_vector_type(16))) _Float16 v16h;
typedef __attribute__((ext_vector_type(8)))  float    v8f;

#define B_   4
#define N_   2048
#define M_   1024
#define K_   64
#define DIN  64
#define H1_  64
#define H2_  64
#define DOUT 128
#define R2_  0.04f

// ---------------- order-preserving float<->uint for atomic max ----------------
__device__ __forceinline__ unsigned enc_f32(float f) {
    unsigned u = __float_as_uint(f);
    return (u & 0x80000000u) ? ~u : (u | 0x80000000u);
}
__device__ __forceinline__ float dec_f32(unsigned u) {
    unsigned b = (u & 0x80000000u) ? (u & 0x7FFFFFFFu) : ~u;
    return __uint_as_float(b);
}
#define ENC_NEG_INF 0x007FFFFFu   // enc(-inf)

// ---------------- 1) farthest point sampling: one block per cloud -------------
__global__ __launch_bounds__(1024) void fps_kernel(const float* __restrict__ pos,
                                                   float* __restrict__ pos_s) {
    const int b = blockIdx.x;
    const int t = threadIdx.x;
    __shared__ float dists[N_];
    __shared__ float rv[1024];
    __shared__ int   ri[1024];
    __shared__ float fpx, fpy, fpz;
    __shared__ int   far;
    const float* pb = pos + (size_t)b * N_ * 3;

    for (int i = t; i < N_; i += 1024) dists[i] = 1e10f;
    if (t == 0) far = 0;
    __syncthreads();

    for (int m = 0; m < M_; ++m) {
        if (t == 0) {
            int f = far;
            fpx = pb[f * 3 + 0]; fpy = pb[f * 3 + 1]; fpz = pb[f * 3 + 2];
            float* o = pos_s + ((size_t)b * M_ + m) * 3;
            o[0] = fpx; o[1] = fpy; o[2] = fpz;
        }
        __syncthreads();
        float bestv = -1.0f; int besti = 0;
        for (int i = t; i < N_; i += 1024) {
            float dx = pb[i * 3 + 0] - fpx;
            float dy = pb[i * 3 + 1] - fpy;
            float dz = pb[i * 3 + 2] - fpz;
            float d  = dx * dx + dy * dy + dz * dz;
            float nd = fminf(dists[i], d);
            dists[i] = nd;
            if (nd > bestv) { bestv = nd; besti = i; }
        }
        rv[t] = bestv; ri[t] = besti;
        __syncthreads();
        for (int s = 512; s > 0; s >>= 1) {
            if (t < s) {
                if (rv[t + s] > rv[t]) { rv[t] = rv[t + s]; ri[t] = ri[t + s]; }
            }
            __syncthreads();
        }
        if (t == 0) far = ri[0];
        __syncthreads();
    }
}

// ---------------- 2) ball query: one thread per source point ------------------
__global__ void ballq_kernel(const float* __restrict__ pos,
                             const float* __restrict__ pos_s,
                             int* __restrict__ tgt) {
    int g = blockIdx.x * blockDim.x + threadIdx.x;
    if (g >= B_ * N_) return;
    int b = g / N_;
    const float* p  = pos + (size_t)g * 3;
    const float px = p[0], py = p[1], pz = p[2];
    const float* ps = pos_s + (size_t)b * M_ * 3;
    int* tg = tgt + (size_t)g * K_;
    int cnt = 0;
    for (int m = 0; m < M_ && cnt < K_; ++m) {      // ascending m = sorted column order
        float dx = px - ps[m * 3 + 0];
        float dy = py - ps[m * 3 + 1];
        float dz = pz - ps[m * 3 + 2];
        if (dx * dx + dy * dy + dz * dz < R2_) tg[cnt++] = m;
    }
    for (int k = cnt; k < K_; ++k) tg[k] = -1;
}

// ---------------- 3) weight transpose to N-major f16 --------------------------
__global__ void prep_kernel(const float* __restrict__ W2, const float* __restrict__ W3,
                            _Float16* __restrict__ W2t, _Float16* __restrict__ W3t) {
    int g = blockIdx.x * blockDim.x + threadIdx.x;
    if (g < H2_ * H1_) {                     // W2t[n][k] = W2[k][n]
        int n = g / H1_, k = g % H1_;
        W2t[g] = (_Float16)W2[k * H2_ + n];
    }
    if (g < DOUT * H2_) {                    // W3t[n][k] = W3[k][n]
        int n = g / H2_, k = g % H2_;
        W3t[g] = (_Float16)W3[k * DOUT + n];
    }
}

__global__ void init_kernel(unsigned* __restrict__ accum) {
    int g = blockIdx.x * blockDim.x + threadIdx.x;
    if (g < B_ * M_ * DOUT) accum[g] = ENC_NEG_INF;
}

// ---------------- 4) per-point MLP with WMMA + scatter max --------------------
// One 4-wave block per source point. Each wave owns a 16-row edge strip.
__global__ __launch_bounds__(128) void mlp_kernel(
    const float* __restrict__ x,   const float* __restrict__ pos,
    const float* __restrict__ pos_s, const int* __restrict__ tgt,
    const float* __restrict__ W1,  const float* __restrict__ b1,
    const _Float16* __restrict__ W2t, const float* __restrict__ b2,
    const _Float16* __restrict__ W3t, const float* __restrict__ b3,
    unsigned* __restrict__ accum)
{
    const int pt   = blockIdx.x;          // 0 .. B*N-1
    const int b    = pt / N_;
    const int t    = threadIdx.x;
    const int lane = t & 31;
    const int wave = t >> 5;
    const int laneHi = lane >> 4;         // 0 (lanes 0-15) or 1 (lanes 16-31)
    const int ln16   = lane & 15;

    __shared__ float    xs[DIN];
    __shared__ float    base[H1_];
    __shared__ float    rel[K_][3];
    __shared__ int      tg[K_];
    __shared__ _Float16 h1[K_][H1_];
    __shared__ _Float16 h2[K_][H2_];

    // hint weights into cache (global_prefetch_b8)
    __builtin_prefetch(W2t + t, 0, 0);
    __builtin_prefetch(W3t + t, 0, 0);

    if (t < DIN) xs[t] = x[(size_t)pt * DIN + t];
    if (t < K_)  tg[t] = tgt[(size_t)pt * K_ + t];
    __syncthreads();

    if (t < K_) {
        int m = tg[t];
        if (m >= 0) {
            const float* ps = pos_s + ((size_t)b * M_ + m) * 3;
            rel[t][0] = pos[(size_t)pt * 3 + 0] - ps[0];
            rel[t][1] = pos[(size_t)pt * 3 + 1] - ps[1];
            rel[t][2] = pos[(size_t)pt * 3 + 2] - ps[2];
        } else {
            rel[t][0] = 0.f; rel[t][1] = 0.f; rel[t][2] = 0.f;
        }
    }
    // layer-1 x-part: base[c] = x . W1[0:64, c] + b1[c]   (shared by all K edges)
    if (t < H1_) {
        float s = b1[t];
        #pragma unroll 8
        for (int r = 0; r < DIN; ++r) s += xs[r] * W1[r * H1_ + t];
        base[t] = s;
    }
    __syncthreads();

    // layer-1 per-edge rank-3 update + relu -> f16
    for (int i = t; i < K_ * H1_; i += 128) {
        int k = i >> 6, c = i & 63;
        float v = base[c]
                + rel[k][0] * W1[(DIN + 0) * H1_ + c]
                + rel[k][1] * W1[(DIN + 1) * H1_ + c]
                + rel[k][2] * W1[(DIN + 2) * H1_ + c];
        h1[k][c] = (_Float16)fmaxf(v, 0.0f);
    }
    __syncthreads();

    const int rowA = (wave << 4) + ln16;  // A-fragment row for this lane

    // ---- layer 2: h2 = relu(h1 @ W2 + b2), 64x64 @ 64x64, 4 tiles/wave ----
    for (int nt = 0; nt < 4; ++nt) {
        v8f acc = {};
        const int col = (nt << 4) + ln16;
        #pragma unroll
        for (int k0 = 0; k0 < H1_; k0 += 32) {
            v16h a, bf;
            const int ka = k0 + (laneHi << 4);
            #pragma unroll
            for (int e = 0; e < 8; ++e) {         // contiguous k order (matches B load)
                a[e]     = h1[rowA][ka + e];
                a[e + 8] = h1[rowA][ka + 8 + e];
            }
            bf = *(const v16h*)(W2t + (size_t)col * H1_ + k0 + (laneHi << 4));
            acc = __builtin_amdgcn_wmma_f32_16x16x32_f16(
                false, a, false, bf, (short)0, acc, false, false);
        }
        #pragma unroll
        for (int j = 0; j < 8; ++j) {             // ISA 32-bit C/D layout
            int r = (wave << 4) + j + (laneHi << 3);
            h2[r][col] = (_Float16)fmaxf(acc[j] + b2[col], 0.0f);
        }
    }
    __syncthreads();

    // ---- layer 3: msg = h2 @ W3 + b3, 64x64 @ 64x128, 8 tiles/wave + scatter ----
    for (int nt = 0; nt < 8; ++nt) {
        v8f acc = {};
        const int col = (nt << 4) + ln16;
        #pragma unroll
        for (int k0 = 0; k0 < H2_; k0 += 32) {
            v16h a, bf;
            const int ka = k0 + (laneHi << 4);
            #pragma unroll
            for (int e = 0; e < 8; ++e) {
                a[e]     = h2[rowA][ka + e];
                a[e + 8] = h2[rowA][ka + 8 + e];
            }
            bf = *(const v16h*)(W3t + (size_t)col * H2_ + k0 + (laneHi << 4));
            acc = __builtin_amdgcn_wmma_f32_16x16x32_f16(
                false, a, false, bf, (short)0, acc, false, false);
        }
        #pragma unroll
        for (int j = 0; j < 8; ++j) {
            int r = (wave << 4) + j + (laneHi << 3);
            int m = tg[r];
            if (m >= 0) {
                float v = acc[j] + b3[col];
                atomicMax(accum + ((size_t)b * M_ + m) * DOUT + col, enc_f32(v));
            }
        }
    }
}

// ---------------- 5) finalize: decode max, emit pos_out / batch_out -----------
__global__ void finalize_kernel(const unsigned* __restrict__ accum,
                                const float* __restrict__ pos_s,
                                float* __restrict__ out) {
    const int T1 = B_ * M_ * DOUT;
    const int T2 = B_ * M_ * 3;
    const int T3 = B_ * M_;
    int g = blockIdx.x * blockDim.x + threadIdx.x;
    if (g < T1) {
        unsigned u = accum[g];
        out[g] = (u == ENC_NEG_INF) ? 0.0f : dec_f32(u);   // empty centroid -> 0
    } else if (g < T1 + T2) {
        out[g] = pos_s[g - T1];
    } else if (g < T1 + T2 + T3) {
        int i = g - T1 - T2;
        ((int*)out)[g] = i / M_;                            // batch_out int32
    }
}

extern "C" void kernel_launch(void* const* d_in, const int* in_sizes, int n_in,
                              void* d_out, int out_size, void* d_ws, size_t ws_size,
                              hipStream_t stream) {
    const float* x   = (const float*)d_in[0];
    const float* pos = (const float*)d_in[1];
    // d_in[2] = batch (unused: layout is implicit)
    const float* W1  = (const float*)d_in[3];
    const float* b1  = (const float*)d_in[4];
    const float* W2  = (const float*)d_in[5];
    const float* b2  = (const float*)d_in[6];
    const float* W3  = (const float*)d_in[7];
    const float* b3  = (const float*)d_in[8];

    // workspace carve-out (256B aligned chunks)
    char* ws = (char*)d_ws;
    size_t o = 0;
    auto take = [&](size_t bytes) { size_t r = o; o += (bytes + 255) & ~(size_t)255; return r; };
    float*    pos_s = (float*)   (ws + take((size_t)B_ * M_ * 3 * sizeof(float)));
    int*      tgt   = (int*)     (ws + take((size_t)B_ * N_ * K_ * sizeof(int)));
    _Float16* W2t   = (_Float16*)(ws + take((size_t)H2_ * H1_ * sizeof(_Float16)));
    _Float16* W3t   = (_Float16*)(ws + take((size_t)DOUT * H2_ * sizeof(_Float16)));
    unsigned* accum = (unsigned*)(ws + take((size_t)B_ * M_ * DOUT * sizeof(unsigned)));

    fps_kernel<<<B_, 1024, 0, stream>>>(pos, pos_s);
    ballq_kernel<<<(B_ * N_ + 255) / 256, 256, 0, stream>>>(pos, pos_s, tgt);
    prep_kernel<<<(DOUT * H2_ + 255) / 256, 256, 0, stream>>>(W2, W3, W2t, W3t);
    init_kernel<<<(B_ * M_ * DOUT + 255) / 256, 256, 0, stream>>>(accum);
    mlp_kernel<<<B_ * N_, 128, 0, stream>>>(x, pos, pos_s, tgt,
                                            W1, b1, W2t, b2, W3t, b3, accum);
    int tot = B_ * M_ * DOUT + B_ * M_ * 3 + B_ * M_;
    finalize_kernel<<<(tot + 255) / 256, 256, 0, stream>>>(accum, pos_s, (float*)d_out);
}
